// Combined_MIL_Feature_Attention_MultiBag_29051158790279
// MI455X (gfx1250) — matrix operand order repaired and checked by reading, not compiled, not run on hardware
//
#include <hip/hip_runtime.h>
#include <hip/hip_bf16.h>
#include <math.h>

// Problem sizes
static constexpr int kB = 256;
static constexpr int kT = 500;
static constexpr int kM = 512;
static constexpr int kL = 128;

typedef __bf16 bf16_t;
typedef __attribute__((ext_vector_type(16))) __bf16 v16bf;
typedef __attribute__((ext_vector_type(8)))  __bf16 v8bf;
typedef __attribute__((ext_vector_type(8)))  float  v8f;
typedef __attribute__((ext_vector_type(4)))  float  f32x4;

// LDS tile pitch (bf16 elements). 520*2B = 1040B row pitch -> lane i hits
// bank (4*i) mod 64: conflict-light for the 16-byte fragment loads.
static constexpr int kPitch = 520;

// ---- fast transcendentals: gfx1250 has native v_tanh_f32 / v_rcp_f32 ------
__device__ __forceinline__ float fast_tanh(float x) {
#if __has_builtin(__builtin_amdgcn_tanhf)
  return __builtin_amdgcn_tanhf(x);          // v_tanh_f32 (TRANS, co-executes)
#else
  return tanhf(x);
#endif
}

__device__ __forceinline__ float fast_rcp(float x) {
#if __has_builtin(__builtin_amdgcn_rcpf)
  return __builtin_amdgcn_rcpf(x);           // v_rcp_f32
#else
  return 1.0f / x;
#endif
}

__device__ __forceinline__ float fast_sigmoid(float x) {
  return fast_rcp(1.0f + __expf(-x));
}

// ---------------------------------------------------------------------------
// Kernel 0: convert Wv, Wu (L x M fp32 row-major) -> bf16 row-major in d_ws.
// ---------------------------------------------------------------------------
__global__ void cvt_weights_kernel(const float* __restrict__ Wv,
                                   const float* __restrict__ Wu,
                                   bf16_t* __restrict__ wvb,
                                   bf16_t* __restrict__ wub) {
  int i = blockIdx.x * blockDim.x + threadIdx.x;   // 0 .. L*M-1 (65536)
  wvb[i] = (bf16_t)Wv[i];
  wub[i] = (bf16_t)Wu[i];
}

// ---------------------------------------------------------------------------
// Kernel 1: fused V=tanh(HWv^T+bv), U=sigmoid(HWu^T+bu),
//           A_pre = (V*U)·Ww + bw, one block per 16 rows of H (B*T rows).
// 8 waves per block; wave w handles L-tile w (16 columns of L=128).
// ---------------------------------------------------------------------------
__global__ __launch_bounds__(256, 2)
void attn_score_kernel(const float* __restrict__ H,
                       const bf16_t* __restrict__ Wvb,
                       const bf16_t* __restrict__ Wub,
                       const float* __restrict__ bv,
                       const float* __restrict__ bu,
                       const float* __restrict__ Ww,
                       const float* __restrict__ bw,
                       float* __restrict__ A_pre_out) {
  __shared__ bf16_t Hs[16 * kPitch];
  __shared__ float  s_row[16];

  const int tid  = threadIdx.x;
  const int wave = tid >> 5;         // 0..7  -> L-tile index
  const int lane = tid & 31;
  const int half = lane >> 4;        // 0: lanes 0-15, 1: lanes 16-31
  const int l16  = lane & 15;
  const long rowBase = (long)blockIdx.x * 16;   // flat row in (B*T)

  // ---- stage 16x512 fp32 tile -> bf16 LDS tile (each thread: 32 elems) ----
  {
    const int r  = tid >> 4;          // 0..15
    const int c0 = (tid & 15) * 32;   // 0..480
    const float* src = H + (rowBase + r) * (long)kM + c0;
    #pragma unroll
    for (int i = 0; i < 4; ++i) {
      f32x4 a = ((const f32x4*)src)[2 * i];
      f32x4 b = ((const f32x4*)src)[2 * i + 1];
      v8bf h;
      #pragma unroll
      for (int j = 0; j < 4; ++j) { h[j] = (bf16_t)a[j]; h[4 + j] = (bf16_t)b[j]; }
      *(v8bf*)&Hs[r * kPitch + c0 + 8 * i] = h;
    }
  }
  if (tid < 16) s_row[tid] = 0.0f;
  __syncthreads();

  // per-lane L column owned by this wave
  const int   lidx = wave * 16 + l16;
  const float bv_l = bv[lidx];
  const float bu_l = bu[lidx];
  const float ww_l = Ww[lidx];
  const float bw0  = bw[0];

  const bf16_t* wvrow = Wvb + (size_t)lidx * kM;  // B-frag: lane n holds row lt*16+n
  const bf16_t* wurow = Wub + (size_t)lidx * kM;

  v8f accV = {};
  v8f accU = {};

  // ---- K loop: 16 steps of K=32 over M=512 ----
  #pragma unroll 4
  for (int ks = 0; ks < 16; ++ks) {
    // A fragment (16x32 bf16): lane half h reads row l16,
    // K-contiguous 16 elems [h*16, h*16+15] (matches B).
    const int mA0 = ks * 32 + half * 16;
    v8bf a0 = *(const v8bf*)&Hs[l16 * kPitch + mA0];
    v8bf a1 = *(const v8bf*)&Hs[l16 * kPitch + mA0 + 8];
    v16bf afrag = __builtin_shufflevector(
        a0, a1, 0, 1, 2, 3, 4, 5, 6, 7, 8, 9, 10, 11, 12, 13, 14, 15);

    // B fragment (32x16 bf16): lane n = column n = weight row lt*16+n,
    // K-contiguous 16 elems, lanes 16-31 take K 16..31.
    const int mB0 = ks * 32 + half * 16;
    v16bf bvf = *(const v16bf*)(wvrow + mB0);
    v16bf buf = *(const v16bf*)(wurow + mB0);

    accV = __builtin_amdgcn_wmma_f32_16x16x32_bf16(
        false, afrag, false, bvf, (short)0, accV, false, false);
    accU = __builtin_amdgcn_wmma_f32_16x16x32_bf16(
        false, afrag, false, buf, (short)0, accU, false, false);
  }

  // ---- epilogue: act(V)*act(U)*Ww, reduce over L ----
  // D layout: vgpr j -> M row (j + 8*half), N col = l16.
  #pragma unroll
  for (int j = 0; j < 8; ++j) {
    float v = fast_tanh(accV[j] + bv_l);
    float u = fast_sigmoid(accU[j] + bu_l);
    float p = v * u * ww_l;
    // sum across the 16 lanes of this half-wave (xor masks < 16 stay in half)
    p += __shfl_xor(p, 1);
    p += __shfl_xor(p, 2);
    p += __shfl_xor(p, 4);
    p += __shfl_xor(p, 8);
    if (l16 == 0) atomicAdd(&s_row[j + 8 * half], p);   // ds_add_f32 across waves
  }
  __syncthreads();

  if (tid < 16) A_pre_out[rowBase + tid] = s_row[tid] + bw0;
}

// ---------------------------------------------------------------------------
// Kernel 2: per-bag softmax over T, attention pooling over T, 2-wide head.
// One block (512 threads) per bag; thread m owns feature column m.
// ---------------------------------------------------------------------------
__global__ __launch_bounds__(512, 2)
void softmax_pool_kernel(const float* __restrict__ H,
                         const float* __restrict__ A_pre,
                         const float* __restrict__ Wc,
                         const float* __restrict__ bc,
                         float* __restrict__ out,
                         float* __restrict__ A_sft) {
  __shared__ float red[512];
  __shared__ float a_s[kT];

  const int b   = blockIdx.x;
  const int tid = threadIdx.x;

  float x = (tid < kT) ? A_pre[(size_t)b * kT + tid] : -3.4e38f;

  // max reduce
  red[tid] = x;
  __syncthreads();
  #pragma unroll
  for (int s = 256; s > 0; s >>= 1) {
    if (tid < s) red[tid] = fmaxf(red[tid], red[tid + s]);
    __syncthreads();
  }
  const float mx = red[0];
  __syncthreads();

  float e = (tid < kT) ? __expf(x - mx) : 0.0f;
  red[tid] = e;
  __syncthreads();
  #pragma unroll
  for (int s = 256; s > 0; s >>= 1) {
    if (tid < s) red[tid] += red[tid + s];
    __syncthreads();
  }
  const float S = red[0];
  __syncthreads();

  const float a = e / S;
  if (tid < kT) {
    a_s[tid] = a;
    A_sft[(size_t)b * kT + tid] = a;
  }

  // bias_rel = sum of softmax weights
  red[tid] = (tid < kT) ? a : 0.0f;
  __syncthreads();
  #pragma unroll
  for (int s = 256; s > 0; s >>= 1) {
    if (tid < s) red[tid] += red[tid + s];
    __syncthreads();
  }
  const float suma = red[0];
  __syncthreads();

  // attention pooling: Mfeat[m] = sum_t a[t] * H[b,t,m]  (coalesced in m)
  const float* Hb = H + (size_t)b * kT * kM + tid;
  float acc = 0.0f;
  #pragma unroll 4
  for (int t = 0; t < kT; t += 4) {           // T = 500, multiple of 4
    acc = fmaf(a_s[t],     Hb[(size_t)(t)     * kM], acc);
    acc = fmaf(a_s[t + 1], Hb[(size_t)(t + 1) * kM], acc);
    acc = fmaf(a_s[t + 2], Hb[(size_t)(t + 2) * kM], acc);
    acc = fmaf(a_s[t + 3], Hb[(size_t)(t + 3) * kM], acc);
  }

  // head: out[b,k] = sum_m Mfeat[m] * Wc[k,m] + bc[k]*suma
  float p0 = acc * Wc[tid];
  float p1 = acc * Wc[kM + tid];
  red[tid] = p0;
  __syncthreads();
  #pragma unroll
  for (int s = 256; s > 0; s >>= 1) {
    if (tid < s) red[tid] += red[tid + s];
    __syncthreads();
  }
  const float S0 = red[0];
  __syncthreads();
  red[tid] = p1;
  __syncthreads();
  #pragma unroll
  for (int s = 256; s > 0; s >>= 1) {
    if (tid < s) red[tid] += red[tid + s];
    __syncthreads();
  }
  const float S1 = red[0];

  if (tid == 0) {
    out[b * 2 + 0] = S0 + bc[0] * suma;
    out[b * 2 + 1] = S1 + bc[1] * suma;
  }
}

// ---------------------------------------------------------------------------
extern "C" void kernel_launch(void* const* d_in, const int* in_sizes, int n_in,
                              void* d_out, int out_size, void* d_ws, size_t ws_size,
                              hipStream_t stream) {
  const float* H  = (const float*)d_in[0];   // (B,T,M)
  const float* Wv = (const float*)d_in[1];   // (L,M)
  const float* bv = (const float*)d_in[2];   // (L,)
  const float* Wu = (const float*)d_in[3];   // (L,M)
  const float* bu = (const float*)d_in[4];   // (L,)
  const float* Ww = (const float*)d_in[5];   // (1,L)
  const float* bw = (const float*)d_in[6];   // (1,)
  const float* Wc = (const float*)d_in[7];   // (2,M)
  const float* bc = (const float*)d_in[8];   // (2,)

  float* out   = (float*)d_out;              // (B,2)            512
  float* A_sft = out + kB * 2;               // (B,T)            128000
  float* A_pre = A_sft + kB * kT;            // (B,1,T)          128000

  bf16_t* wvb = (bf16_t*)d_ws;               // 64K bf16
  bf16_t* wub = wvb + (size_t)kL * kM;       // 64K bf16

  cvt_weights_kernel<<<(kL * kM) / 256, 256, 0, stream>>>(Wv, Wu, wvb, wub);

  const int nTiles = (kB * kT) / 16;         // 128000/16 = 8000
  attn_score_kernel<<<nTiles, 256, 0, stream>>>(H, wvb, wub, bv, bu, Ww, bw, A_pre);

  softmax_pool_kernel<<<kB, 512, 0, stream>>>(H, A_pre, Wc, bc, out, A_sft);
}
